// CrossAttention_7026566496881
// MI455X (gfx1250) — matrix operand order repaired and checked by reading, not compiled, not run on hardware
//
#include <hip/hip_runtime.h>
#include <hip/hip_bf16.h>

// ---------------------------------------------------------------------------
// CDNA5 (gfx1250) fused cross-attention, bf16 WMMA path, f32 accumulate.
// wave32 only; all matmuls via v_wmma_f32_16x16x32_bf16.
// ---------------------------------------------------------------------------

typedef __bf16 bf16;
typedef __attribute__((ext_vector_type(16))) __bf16 v16bf;
typedef __attribute__((ext_vector_type(8)))  __bf16 v8bf;
typedef __attribute__((ext_vector_type(8)))  float  v8f;

#define DIM   1024
#define SEQ   2048
#define BATCH 8
#define ROWS_TOTAL (BATCH * SEQ)         // 16384
#define ATT_SCALE 0.08838834764831845f   // 128^-0.5

// ---------------------------------------------------------------------------
// WMMA helper: D = A(16x32 bf16) * B(32x16 bf16) + C(16x16 f32)
// ---------------------------------------------------------------------------
__device__ __forceinline__ v8f wmma_bf16(v16bf a, v16bf b, v8f c) {
  return __builtin_amdgcn_wmma_f32_16x16x32_bf16(
      /*neg_a=*/false, a, /*neg_b=*/false, b,
      /*c_mod=*/(short)0, c, /*reuse_a=*/false, /*reuse_b=*/false);
}

// A-matrix fragment (16x32, K-major source row `row`):
// lanes 0-15 (half=0): K = {0..7, 16..23}; lanes 16-31 (half=1): K = {8..15, 24..31}
__device__ __forceinline__ v16bf frag_a(const bf16* row, int k, int half) {
  v8bf lo = *(const v8bf*)(row + k + 8 * half);
  v8bf hi = *(const v8bf*)(row + k + 16 + 8 * half);
  v16bf a;
#pragma unroll
  for (int i = 0; i < 8; ++i) { a[i] = lo[i]; a[i + 8] = hi[i]; }
  return a;
}

// B-matrix fragment (32x16): lane = column, K = {0..7,16..23} (half=0) / {8..15,24..31} (half=1)
__device__ __forceinline__ v16bf frag_b(const bf16* row, int k, int half) {
  v8bf lo = *(const v8bf*)(row + k + 8 * half);
  v8bf hi = *(const v8bf*)(row + k + 16 + 8 * half);
  v16bf b;
#pragma unroll
  for (int i = 0; i < 8; ++i) { b[i] = lo[i]; b[i + 8] = hi[i]; }
  return b;
}

__device__ __forceinline__ float wave_max(float v) {
#pragma unroll
  for (int o = 16; o > 0; o >>= 1) v = fmaxf(v, __shfl_xor(v, o, 32));
  return v;
}
__device__ __forceinline__ float wave_sum(float v) {
#pragma unroll
  for (int o = 16; o > 0; o >>= 1) v += __shfl_xor(v, o, 32);
  return v;
}

// ---------------------------------------------------------------------------
// Kernel 0: f32 -> bf16 conversion (n multiple of 4)
// ---------------------------------------------------------------------------
__global__ __launch_bounds__(256) void cvt_f32_bf16(const float* __restrict__ in,
                                                    bf16* __restrict__ out, int n) {
  int i = (blockIdx.x * 256 + threadIdx.x) * 4;
  if (i < n) {
    float4 v = *(const float4*)(in + i);
    out[i + 0] = (bf16)v.x;
    out[i + 1] = (bf16)v.y;
    out[i + 2] = (bf16)v.z;
    out[i + 3] = (bf16)v.w;
  }
}

// ---------------------------------------------------------------------------
// Kernel 1: QKV projection.  C = A * W^T (A:[16384,1024] bf16, W:[1024,1024])
// grid = (4, 128, 3); z=0: Q from x, z=1: K from support, z=2: V from support
// Q,K stored row-major bf16; V stored transposed per batch: Vt[b][d][m].
// WG tile 128x256; 8 waves (2x4), each computes 64x64 (4x4 WMMA frags):
// 16 WMMAs per 8 fragment loads per k-step.
// ---------------------------------------------------------------------------
__global__ __launch_bounds__(256) void gemm_qkv(const bf16* __restrict__ xb,
                                                const bf16* __restrict__ sb,
                                                const bf16* __restrict__ wb,
                                                bf16* __restrict__ Qb,
                                                bf16* __restrict__ Kb,
                                                bf16* __restrict__ Vtb) {
  const int z = blockIdx.z;
  const bf16* A = (z == 0) ? xb : sb;
  const bf16* W = wb + (size_t)z * DIM * DIM;

  const int tid  = threadIdx.x;
  const int lane = tid & 31, wave = tid >> 5;
  const int half = lane >> 4, l16 = lane & 15;
  const int waveM = wave >> 2, waveN = wave & 3;
  const int rowBase = blockIdx.y * 128 + waveM * 64;
  const int colBase = blockIdx.x * 256 + waveN * 64;

  v8f acc[4][4];
#pragma unroll
  for (int i = 0; i < 4; ++i)
#pragma unroll
    for (int j = 0; j < 4; ++j)
#pragma unroll
      for (int e = 0; e < 8; ++e) acc[i][j][e] = 0.0f;

  for (int k = 0; k < DIM; k += 32) {
    v16bf a[4], b[4];
#pragma unroll
    for (int i = 0; i < 4; ++i)
      a[i] = frag_a(A + (size_t)(rowBase + i * 16 + l16) * DIM, k, half);
#pragma unroll
    for (int j = 0; j < 4; ++j)
      b[j] = frag_b(W + (size_t)(colBase + j * 16 + l16) * DIM, k, half);
#pragma unroll
    for (int i = 0; i < 4; ++i)
#pragma unroll
      for (int j = 0; j < 4; ++j) acc[i][j] = wmma_bf16(a[i], b[j], acc[i][j]);
  }

  if (z < 2) {
    bf16* O = (z == 0) ? Qb : Kb;
#pragma unroll
    for (int i = 0; i < 4; ++i)
#pragma unroll
      for (int j = 0; j < 4; ++j)
#pragma unroll
        for (int r = 0; r < 8; ++r) {
          int row = rowBase + i * 16 + r + 8 * half;
          int col = colBase + j * 16 + l16;
          O[(size_t)row * DIM + col] = (bf16)acc[i][j][r];
        }
  } else {  // V transposed per batch: Vt[b][d][m]
#pragma unroll
    for (int i = 0; i < 4; ++i)
#pragma unroll
      for (int j = 0; j < 4; ++j)
#pragma unroll
        for (int r = 0; r < 8; ++r) {
          int g = rowBase + i * 16 + r + 8 * half;   // b*2048 + m
          int bIdx = g >> 11, m = g & (SEQ - 1);
          int d = colBase + j * 16 + l16;
          Vtb[(size_t)bIdx * DIM * SEQ + (size_t)d * SEQ + m] = (bf16)acc[i][j][r];
        }
  }
}

// ---------------------------------------------------------------------------
// Kernel 2: flash attention (single head, d = 1024, mask after softmax).
// grid = 1024 WGs, each handles 16 q-rows of one batch; m-tiles of 128.
// 8 waves: jointly compute S (16x128) -> LDS; online softmax; each wave owns
// a 128-wide d-slice of O.  Epilogue folds the swapaxes/reshape scramble.
// global_prefetch warms the next m-tile's K/V rows while WMMAs run.
// ---------------------------------------------------------------------------
__global__ __launch_bounds__(256) void flash_attn(const bf16* __restrict__ Qb,
                                                  const bf16* __restrict__ Kb,
                                                  const bf16* __restrict__ Vtb,
                                                  const int* __restrict__ mask,
                                                  bf16* __restrict__ Asc) {
  __shared__ __align__(16) bf16  Qs[16 * DIM];   // 32 KB
  __shared__ float               Ss[16 * 128];   // 8 KB
  __shared__ __align__(16) bf16  Ps[16 * 128];   // 4 KB
  __shared__ float rowM[16], rowS[16], rowScale[16];

  const int tid  = threadIdx.x;
  const int lane = tid & 31, wave = tid >> 5;
  const int half = lane >> 4, l16 = lane & 15;
  const int rowBase = blockIdx.x * 16;          // global n-row (b*2048+n)
  const int batch   = rowBase >> 11;

  // stage Q tile (16x1024 bf16) into LDS
  {
    int r = tid >> 4, cb = (tid & 15) * 64;
#pragma unroll
    for (int i = 0; i < 8; ++i)
      *(v8bf*)&Qs[r * DIM + cb + i * 8] =
          *(const v8bf*)&Qb[(size_t)(rowBase + r) * DIM + cb + i * 8];
  }
  if (tid < 16) { rowM[tid] = -1e30f; rowS[tid] = 0.0f; }
  __syncthreads();

  const bf16* Kbase = Kb  + (size_t)batch * SEQ * DIM;
  const bf16* Vbase = Vtb + (size_t)batch * DIM * SEQ;

  v8f o[8];
#pragma unroll
  for (int f = 0; f < 8; ++f)
#pragma unroll
    for (int e = 0; e < 8; ++e) o[f][e] = 0.0f;

  for (int mt = 0; mt < SEQ; mt += 128) {
    // prefetch next m-tile's K rows and V columns into cache (gfx1250 path)
    if (mt + 128 < SEQ) {
      __builtin_prefetch(Kbase + (size_t)(mt + 128 + wave * 16 + l16) * DIM +
                             64 * half, 0, 1);
      __builtin_prefetch(Vbase + (size_t)(wave * 128 + lane * 4) * SEQ +
                             (mt + 128), 0, 1);
    }

    // ---- S = Q * K^T : wave computes columns [wave*16, wave*16+16) ----
    v8f s;
#pragma unroll
    for (int e = 0; e < 8; ++e) s[e] = 0.0f;
    for (int k = 0; k < DIM; k += 32) {
      v16bf a = frag_a(Qs + (size_t)l16 * DIM, k, half);
      v16bf b = frag_b(Kbase + (size_t)(mt + wave * 16 + l16) * DIM, k, half);
      s = wmma_bf16(a, b, s);
    }
#pragma unroll
    for (int r = 0; r < 8; ++r)
      Ss[(r + 8 * half) * 128 + wave * 16 + l16] = s[r];
    __syncthreads();

    // ---- online softmax: wave handles rows 2*wave, 2*wave+1 ----
#pragma unroll
    for (int rr = 0; rr < 2; ++rr) {
      int row = wave * 2 + rr;
      float sv[4];
#pragma unroll
      for (int c = 0; c < 4; ++c)
        sv[c] = Ss[row * 128 + lane * 4 + c] * ATT_SCALE;
      float mx = fmaxf(fmaxf(sv[0], sv[1]), fmaxf(sv[2], sv[3]));
      mx = wave_max(mx);
      float oldM = rowM[row];
      float newM = fmaxf(oldM, mx);
      float sc   = __expf(oldM - newM);
      float local = 0.0f;
#pragma unroll
      for (int c = 0; c < 4; ++c) {
        float p = __expf(sv[c] - newM);
        local += p;                                  // denominator is UNmasked
        int mk = mask[mt + lane * 4 + c];            // mask applied post-softmax
        Ps[row * 128 + lane * 4 + c] = (bf16)(mk ? p : 0.0f);
      }
      local = wave_sum(local);
      if (lane == 0) {
        rowS[row] = rowS[row] * sc + local;
        rowM[row] = newM;
        rowScale[row] = sc;
      }
    }
    __syncthreads();

    // ---- O = O*scale + P * V : wave owns d-cols [wave*128, wave*128+128) ----
    float scv[8];
#pragma unroll
    for (int r = 0; r < 8; ++r) scv[r] = rowScale[r + 8 * half];
#pragma unroll
    for (int f = 0; f < 8; ++f)
#pragma unroll
      for (int r = 0; r < 8; ++r) o[f][r] *= scv[r];

    for (int kk = 0; kk < 128; kk += 32) {
      v16bf a = frag_a(Ps + (size_t)l16 * 128, kk, half);
#pragma unroll
      for (int f = 0; f < 8; ++f) {
        v16bf b = frag_b(Vbase + (size_t)(wave * 128 + f * 16 + l16) * SEQ + mt,
                         kk, half);
        o[f] = wmma_bf16(a, b, o[f]);
      }
    }
  }

  // ---- epilogue: normalize + scrambled store  out[b,n,d] -> A[b, 2d+n/1024, n%1024]
  float inv[8];
#pragma unroll
  for (int r = 0; r < 8; ++r) inv[r] = 1.0f / rowS[r + 8 * half];
#pragma unroll
  for (int f = 0; f < 8; ++f) {
    int d = wave * 128 + f * 16 + l16;
#pragma unroll
    for (int r = 0; r < 8; ++r) {
      int g = rowBase + r + 8 * half;          // b*2048 + n
      int bIdx = g >> 11, n = g & (SEQ - 1);
      float val = o[f][r] * inv[r];
      size_t idx = (size_t)bIdx * SEQ * DIM +
                   (size_t)(2 * d + (n >> 10)) * DIM + (n & (DIM - 1));
      Asc[idx] = (bf16)val;
    }
  }
}

// ---------------------------------------------------------------------------
// Kernel 3: output projection.  Y = A * W^T + bias, f32 output.
// Same 64x64-per-wave tiling as gemm_qkv.
// ---------------------------------------------------------------------------
__global__ __launch_bounds__(256) void gemm_proj(const bf16* __restrict__ A,
                                                 const bf16* __restrict__ W,
                                                 const float* __restrict__ bias,
                                                 float* __restrict__ out) {
  const int tid  = threadIdx.x;
  const int lane = tid & 31, wave = tid >> 5;
  const int half = lane >> 4, l16 = lane & 15;
  const int waveM = wave >> 2, waveN = wave & 3;
  const int rowBase = blockIdx.y * 128 + waveM * 64;
  const int colBase = blockIdx.x * 256 + waveN * 64;

  v8f acc[4][4];
#pragma unroll
  for (int i = 0; i < 4; ++i)
#pragma unroll
    for (int j = 0; j < 4; ++j)
#pragma unroll
      for (int e = 0; e < 8; ++e) acc[i][j][e] = 0.0f;

  for (int k = 0; k < DIM; k += 32) {
    v16bf a[4], b[4];
#pragma unroll
    for (int i = 0; i < 4; ++i)
      a[i] = frag_a(A + (size_t)(rowBase + i * 16 + l16) * DIM, k, half);
#pragma unroll
    for (int j = 0; j < 4; ++j)
      b[j] = frag_b(W + (size_t)(colBase + j * 16 + l16) * DIM, k, half);
#pragma unroll
    for (int i = 0; i < 4; ++i)
#pragma unroll
      for (int j = 0; j < 4; ++j) acc[i][j] = wmma_bf16(a[i], b[j], acc[i][j]);
  }

#pragma unroll
  for (int i = 0; i < 4; ++i)
#pragma unroll
    for (int j = 0; j < 4; ++j)
#pragma unroll
      for (int r = 0; r < 8; ++r) {
        int row = rowBase + i * 16 + r + 8 * half;
        int col = colBase + j * 16 + l16;
        out[(size_t)row * DIM + col] = acc[i][j][r] + bias[col];
      }
}

// ---------------------------------------------------------------------------
// launch
// ---------------------------------------------------------------------------
extern "C" void kernel_launch(void* const* d_in, const int* in_sizes, int n_in,
                              void* d_out, int out_size, void* d_ws, size_t ws_size,
                              hipStream_t stream) {
  const float* x       = (const float*)d_in[0];
  const float* support = (const float*)d_in[1];
  const int*   mask    = (const int*)d_in[2];
  const float* qkv_w   = (const float*)d_in[3];
  const float* proj_w  = (const float*)d_in[4];
  const float* proj_b  = (const float*)d_in[5];
  float* out = (float*)d_out;

  // workspace layout (bytes), total = 200 MB
  char* ws = (char*)d_ws;
  const size_t SZ_ACT = (size_t)ROWS_TOTAL * DIM * sizeof(bf16);  // 32 MB
  bf16* xb     = (bf16*)(ws);
  bf16* sb     = (bf16*)(ws + SZ_ACT);
  bf16* wqkvb  = (bf16*)(ws + 2 * SZ_ACT);
  bf16* wprojb = (bf16*)(ws + 2 * SZ_ACT + (size_t)3 * DIM * DIM * sizeof(bf16));
  bf16* Qb     = (bf16*)(ws + 2 * SZ_ACT + (size_t)4 * DIM * DIM * sizeof(bf16));
  bf16* Kb     = Qb  + (size_t)ROWS_TOTAL * DIM;
  bf16* Vtb    = Kb  + (size_t)ROWS_TOTAL * DIM;
  bf16* Asc    = Vtb + (size_t)ROWS_TOTAL * DIM;

  // 0) convert to bf16
  int nAct = ROWS_TOTAL * DIM;                  // 16777216
  cvt_f32_bf16<<<nAct / 1024, 256, 0, stream>>>(x, xb, nAct);
  cvt_f32_bf16<<<nAct / 1024, 256, 0, stream>>>(support, sb, nAct);
  int nW = 3 * DIM * DIM;
  cvt_f32_bf16<<<nW / 1024, 256, 0, stream>>>(qkv_w, wqkvb, nW);
  int nP = DIM * DIM;
  cvt_f32_bf16<<<nP / 1024, 256, 0, stream>>>(proj_w, wprojb, nP);

  // 1) Q/K/V projections (V stored transposed per batch)
  gemm_qkv<<<dim3(DIM / 256, ROWS_TOTAL / 128, 3), 256, 0, stream>>>(
      xb, sb, wqkvb, Qb, Kb, Vtb);

  // 2) flash attention -> scrambled activations
  flash_attn<<<dim3(ROWS_TOTAL / 16), 256, 0, stream>>>(Qb, Kb, Vtb, mask, Asc);

  // 3) output projection + bias
  gemm_proj<<<dim3(DIM / 256, ROWS_TOTAL / 128), 256, 0, stream>>>(
      Asc, wprojb, proj_b, out);
}
